// TimeConditionedAttention_10161892623050
// MI455X (gfx1250) — matrix operand order repaired and checked by reading, not compiled, not run on hardware
//
#include <hip/hip_runtime.h>
#include <hip/hip_bf16.h>

typedef __attribute__((ext_vector_type(16))) _Float16 v16h;
typedef __attribute__((ext_vector_type(8)))  float    v8f;
typedef __attribute__((ext_vector_type(4)))  unsigned int u32x4;
typedef __attribute__((ext_vector_type(8)))  int i32x8;
typedef __attribute__((ext_vector_type(4)))  int i32x4;

#define NB   4
#define NC   64
#define NSP  4096   // H*W
#define NTOT (NB * NSP)

// ---------------------------------------------------------------------------
// Shared WMMA helpers (16x16x32 f16 -> f32), layouts per CDNA5 ISA:
//   A 16x32 f16 : row = lane&15; a[i<8] K = ks+8*half+i; a[i>=8] K = ks+16+8*half+(i-8)
//   B 32x16 f16 : col = lane&15; b[i<8] K = ks+8*half+i; b[i>=8] K = ks+16+8*half+(i-8)
//   C/D 16x16 f32: row = r + 8*half; col = lane&15
// ---------------------------------------------------------------------------
__device__ __forceinline__ v8f wmma2(const v16h& a0, const v16h& a1,
                                     const v16h& b0, const v16h& b1, v8f c) {
  c = __builtin_amdgcn_wmma_f32_16x16x32_f16(false, a0, false, b0,
                                             (short)0, c, false, false);
  c = __builtin_amdgcn_wmma_f32_16x16x32_f16(false, a1, false, b1,
                                             (short)0, c, false, false);
  return c;
}

__device__ __forceinline__ void load_wB(const float* __restrict__ w, int col,
                                        int half, v16h& b0, v16h& b1) {
#pragma unroll
  for (int i = 0; i < 16; ++i) {
    b0[i] = (_Float16)w[(col << 6) + (i >> 3) * 16 + 8 * half + (i & 7)];
    b1[i] = (_Float16)w[(col << 6) + 32 + (i >> 3) * 16 + 8 * half + (i & 7)];
  }
}

// ---------------------------------------------------------------------------
// Tensor Data Mover: issue one 2-D (or 1-D when tile_d1==0) f16 tile load
// Global -> LDS.  D# bit packing per CDNA5 ISA:
//   group0: [1:0]=count=1, [63:32]=lds_addr, [120:64]=global_addr, [127:126]=2
//   group1: [17:16]=data_size(1 => 2B), [79:48]=tensor_dim0, [111:80]=tensor_dim1,
//           [127:112]=tile_dim0, [143:128]=tile_dim1, [207:160]=tensor_dim0_stride
// Tracked with TENSORcnt (wave-private); completion via s_wait_tensorcnt.
// ---------------------------------------------------------------------------
__device__ __forceinline__ void tdm_load_f16(unsigned lds_off, const void* gaddr,
                                             unsigned tile_d0, unsigned tile_d1,
                                             unsigned tensor_d0, unsigned tensor_d1,
                                             unsigned stride0) {
  const unsigned long long ga = (unsigned long long)(uintptr_t)gaddr;
  u32x4 g0;
  g0[0] = 1u;                                            // count=1, user D#
  g0[1] = lds_off;                                       // LDS byte address
  g0[2] = (unsigned)(ga & 0xFFFFFFFFu);                  // global addr [31:0]
  g0[3] = (unsigned)((ga >> 32) & 0x01FFFFFFu) | 0x80000000u; // [56:32] | type=2
  i32x8 g1;
  g1[0] = (int)(1u << 16);                               // data_size=2B; mask=0
  g1[1] = (int)((tensor_d0 & 0xFFFFu) << 16);            // tensor_dim0 lo16
  g1[2] = (int)((tensor_d0 >> 16) | ((tensor_d1 & 0xFFFFu) << 16));
  g1[3] = (int)((tensor_d1 >> 16) | (tile_d0 << 16));    // tile_dim0
  g1[4] = (int)tile_d1;                                  // tile_dim1 (0 => 1-D)
  g1[5] = (int)stride0;                                  // dim0_stride lo32
  g1[6] = 0;
  g1[7] = 0;
  i32x4 gz = {0, 0, 0, 0};
#if defined(__clang_major__) && __clang_major__ >= 23
  i32x8 gz8 = {0, 0, 0, 0, 0, 0, 0, 0};
  __builtin_amdgcn_tensor_load_to_lds(g0, g1, gz, gz, gz8, 0);
#else
  __builtin_amdgcn_tensor_load_to_lds(g0, g1, gz, gz, 0);
#endif
}

// ---------------------------------------------------------------------------
// Kernel 1: Q/K/V 1x1-conv projection on the WMMA units.
//   Qh, Kh: [B][N][C] f16 (Q pre-scaled by 1/sqrt(C) = 0.125)
//   Vt:     [B][C][N] f16 (transposed)
// ---------------------------------------------------------------------------
__global__ void __launch_bounds__(128)
qkv_project(const float* __restrict__ latent,
            const float* __restrict__ wq, const float* __restrict__ bq,
            const float* __restrict__ wk, const float* __restrict__ bk,
            const float* __restrict__ wv, const float* __restrict__ bv,
            _Float16* __restrict__ Qh, _Float16* __restrict__ Kh,
            _Float16* __restrict__ Vt) {
  const int lane = threadIdx.x & 31;
  const int wid  = threadIdx.x >> 5;
  const int tile = blockIdx.x * 4 + wid;        // 0..1023
  const int b    = tile >> 8;
  const int n0   = (tile & 255) << 4;

  const int half = lane >> 4;
  const int l15  = lane & 15;

  const float* lb = latent + ((size_t)b << 18) + n0 + l15;
  v16h ax0, ax1;
#pragma unroll
  for (int i = 0; i < 8; ++i) {
    ax0[i]     = (_Float16)lb[(size_t)( 0 + 8 * half + i) << 12];
    ax0[i + 8] = (_Float16)lb[(size_t)(16 + 8 * half + i) << 12];
    ax1[i]     = (_Float16)lb[(size_t)(32 + 8 * half + i) << 12];
    ax1[i + 8] = (_Float16)lb[(size_t)(48 + 8 * half + i) << 12];
  }

#pragma unroll
  for (int u = 0; u < 4; ++u) {
    const int col = u * 16 + l15;
    v16h b0, b1;
    v8f c;

    load_wB(wq, col, half, b0, b1);
    c = (v8f){};
    c = wmma2(ax0, ax1, b0, b1, c);
#pragma unroll
    for (int r = 0; r < 8; ++r) {
      const int row = r + 8 * half;
      Qh[(((size_t)(b << 12) + n0 + row) << 6) + col] =
          (_Float16)((c[r] + bq[col]) * 0.125f);
    }

    load_wB(wk, col, half, b0, b1);
    c = (v8f){};
    c = wmma2(ax0, ax1, b0, b1, c);
#pragma unroll
    for (int r = 0; r < 8; ++r) {
      const int row = r + 8 * half;
      Kh[(((size_t)(b << 12) + n0 + row) << 6) + col] =
          (_Float16)(c[r] + bk[col]);
    }

    load_wB(wv, col, half, b0, b1);
    c = (v8f){};
    c = wmma2(ax0, ax1, b0, b1, c);
#pragma unroll
    for (int r = 0; r < 8; ++r) {
      const int row = r + 8 * half;
      Vt[(((size_t)(b << 6) + col) << 12) + n0 + row] =
          (_Float16)(c[r] + bv[col]);
    }
  }
}

// ---------------------------------------------------------------------------
// Kernel 2: fused blockwise attention, flash-style.
// 4 waves/block share one batch; K/V segments (8 KB each) are DMA'd into LDS
// by the Tensor Data Mover (double-buffered, TENSORcnt-tracked) so the 4
// waves read them via ds_load instead of 4x-duplicated global loads.
// ---------------------------------------------------------------------------
__global__ void __launch_bounds__(128)
attn_kernel(const _Float16* __restrict__ Qh,
            const _Float16* __restrict__ Kh,
            const _Float16* __restrict__ Vt,
            float* __restrict__ Obuf) {
  __shared__ _Float16 kbuf[2][64 * 64];   // [key][chan], 8 KB per buffer
  __shared__ _Float16 vbuf[2][64 * 64];   // [chan][key], 8 KB per buffer
  __shared__ _Float16 pbuf[4][16 * 64];   // per-wave P transpose scratch

  const int lane = threadIdx.x & 31;
  const int wid  = threadIdx.x >> 5;
  const int tile = blockIdx.x * 4 + wid;        // 0..1023
  const int b    = tile >> 8;                   // same b for all 4 waves
  const int q0   = (tile & 255) << 4;

  const int half = lane >> 4;
  const int l15  = lane & 15;

  // ---- Q as WMMA A-matrix (two k-steps over C=64).
  const _Float16* qrow = Qh + (((size_t)(b << 12) + q0 + l15) << 6);
  v16h aq0, aq1;
#pragma unroll
  for (int i = 0; i < 8; ++i) {
    aq0[i]     = qrow[ 0 + 8 * half + i];
    aq0[i + 8] = qrow[16 + 8 * half + i];
    aq1[i]     = qrow[32 + 8 * half + i];
    aq1[i + 8] = qrow[48 + 8 * half + i];
  }

  // ---- prologue: TDM seg 0 into buffer 0 (wave 0 only; EXEC ignored by TDM).
  if (wid == 0) {
    tdm_load_f16((unsigned)(uintptr_t)&kbuf[0][0],
                 Kh + (((size_t)(b << 12)) << 6),
                 4096, 0, 4096, 1, 4096);                 // 1-D contiguous 8 KB
    tdm_load_f16((unsigned)(uintptr_t)&vbuf[0][0],
                 Vt + ((size_t)b << 18),
                 64, 64, 4096, 64, 4096);                 // 2-D 64x64, stride 4096
  }

  v8f oacc[4] = {};
  _Float16* pb = pbuf[wid];

  for (int seg = 0; seg < 64; ++seg) {
    const int cur = seg & 1;

    if (wid == 0) __builtin_amdgcn_s_wait_tensorcnt(0);   // seg's DMA complete
    __syncthreads();                                      // publish LDS to block

    if (wid == 0 && seg < 63) {                           // prefetch next segment
      const int nkb = (seg + 1) << 6;
      tdm_load_f16((unsigned)(uintptr_t)&kbuf[cur ^ 1][0],
                   Kh + (((size_t)(b << 12) + nkb) << 6),
                   4096, 0, 4096, 1, 4096);
      tdm_load_f16((unsigned)(uintptr_t)&vbuf[cur ^ 1][0],
                   Vt + ((size_t)b << 18) + nkb,
                   64, 64, 4096, 64, 4096);
    }

    // ---- S = (Q/8) @ K^T : 4 key tiles, B-matrix from LDS.
    v8f s[4];
#pragma unroll
    for (int t = 0; t < 4; ++t) {
      const _Float16* krow = &kbuf[cur][(t * 16 + l15) << 6];
      v16h bk0, bk1;
#pragma unroll
      for (int i = 0; i < 16; ++i) {
        bk0[i] = krow[     (i >> 3) * 16 + 8 * half + (i & 7)];
        bk1[i] = krow[32 + (i >> 3) * 16 + 8 * half + (i & 7)];
      }
      v8f c = {};
      c = wmma2(aq0, aq1, bk0, bk1, c);
      s[t] = c;
    }

    // ---- softmax over the 64 keys of this segment, per query row.
#pragma unroll
    for (int r = 0; r < 8; ++r) {
      float m = fmaxf(fmaxf(s[0][r], s[1][r]), fmaxf(s[2][r], s[3][r]));
#pragma unroll
      for (int mk = 1; mk < 16; mk <<= 1) m = fmaxf(m, __shfl_xor(m, mk, 32));
      float p0 = __expf(s[0][r] - m);
      float p1 = __expf(s[1][r] - m);
      float p2 = __expf(s[2][r] - m);
      float p3 = __expf(s[3][r] - m);
      float l = p0 + p1 + p2 + p3;
#pragma unroll
      for (int mk = 1; mk < 16; mk <<= 1) l += __shfl_xor(l, mk, 32);
      const float inv = __builtin_amdgcn_rcpf(l);
      const int row = r + 8 * half;
      pb[(row << 6) +  0 + l15] = (_Float16)(p0 * inv);
      pb[(row << 6) + 16 + l15] = (_Float16)(p1 * inv);
      pb[(row << 6) + 32 + l15] = (_Float16)(p2 * inv);
      pb[(row << 6) + 48 + l15] = (_Float16)(p3 * inv);
    }

    // CDNA5 split counter: drain DS stores before re-reading P in A-layout.
    asm volatile("s_wait_dscnt 0" ::: "memory");

    // ---- P as WMMA A-matrix (reduction over 64 keys -> 2 k-steps).
    v16h pa0, pa1;
    const _Float16* prow = pb + (l15 << 6);
#pragma unroll
    for (int i = 0; i < 8; ++i) {
      pa0[i]     = prow[ 0 + 8 * half + i];
      pa0[i + 8] = prow[16 + 8 * half + i];
      pa1[i]     = prow[32 + 8 * half + i];
      pa1[i + 8] = prow[48 + 8 * half + i];
    }

    // ---- O += P @ V_seg : 4 channel tiles, B-matrix from LDS.
#pragma unroll
    for (int u = 0; u < 4; ++u) {
      const _Float16* vcol = &vbuf[cur][(u * 16 + l15) << 6];
      v16h bv0, bv1;
#pragma unroll
      for (int i = 0; i < 16; ++i) {
        bv0[i] = vcol[     (i >> 3) * 16 + 8 * half + (i & 7)];
        bv1[i] = vcol[32 + (i >> 3) * 16 + 8 * half + (i & 7)];
      }
      oacc[u] = wmma2(pa0, pa1, bv0, bv1, oacc[u]);
    }

    __syncthreads();   // all waves done with kbuf/vbuf[cur] before TDM reuses it
  }

  // ---- write O tile: [B][N][C] f32.
  float* obase = Obuf + (((size_t)(b << 12) + q0) << 6);
#pragma unroll
  for (int u = 0; u < 4; ++u)
#pragma unroll
    for (int r = 0; r < 8; ++r) {
      const int row = r + 8 * half;
      obase[(row << 6) + u * 16 + l15] = oacc[u][r];
    }
}

// ---------------------------------------------------------------------------
// Kernel 3: output 1x1 conv on the WMMA units. out [B][C][H][W] f32.
// ---------------------------------------------------------------------------
__global__ void __launch_bounds__(128)
out_project(const float* __restrict__ Obuf, const float* __restrict__ wo,
            const float* __restrict__ bo, float* __restrict__ out) {
  const int lane = threadIdx.x & 31;
  const int wid  = threadIdx.x >> 5;
  const int tile = blockIdx.x * 4 + wid;
  const int b    = tile >> 8;
  const int n0   = (tile & 255) << 4;

  const int half = lane >> 4;
  const int l15  = lane & 15;

  const float* orow = Obuf + (((size_t)(b << 12) + n0 + l15) << 6);
  v16h ax0, ax1;
#pragma unroll
  for (int i = 0; i < 8; ++i) {
    ax0[i]     = (_Float16)orow[ 0 + 8 * half + i];
    ax0[i + 8] = (_Float16)orow[16 + 8 * half + i];
    ax1[i]     = (_Float16)orow[32 + 8 * half + i];
    ax1[i + 8] = (_Float16)orow[48 + 8 * half + i];
  }

#pragma unroll
  for (int u = 0; u < 4; ++u) {
    const int col = u * 16 + l15;
    v16h b0, b1;
    load_wB(wo, col, half, b0, b1);
    v8f c = {};
    c = wmma2(ax0, ax1, b0, b1, c);
#pragma unroll
    for (int r = 0; r < 8; ++r) {
      const int row = r + 8 * half;
      out[(((size_t)(b << 6) + col) << 12) + n0 + row] = c[r] + bo[col];
    }
  }
}

// ---------------------------------------------------------------------------
extern "C" void kernel_launch(void* const* d_in, const int* in_sizes, int n_in,
                              void* d_out, int out_size, void* d_ws, size_t ws_size,
                              hipStream_t stream) {
  const float* latent = (const float*)d_in[0];
  // d_in[1] = time (time-MLP is computed-but-unused in the reference; skip)
  const float* wq = (const float*)d_in[2];
  const float* bq = (const float*)d_in[3];
  const float* wk = (const float*)d_in[4];
  const float* bk = (const float*)d_in[5];
  const float* wv = (const float*)d_in[6];
  const float* bv = (const float*)d_in[7];
  const float* wo = (const float*)d_in[8];
  const float* bo = (const float*)d_in[9];

  char* ws = (char*)d_ws;
  _Float16* Qh = (_Float16*)ws;  ws += (size_t)NTOT * NC * sizeof(_Float16); // 2 MB
  _Float16* Kh = (_Float16*)ws;  ws += (size_t)NTOT * NC * sizeof(_Float16); // 2 MB
  _Float16* Vt = (_Float16*)ws;  ws += (size_t)NTOT * NC * sizeof(_Float16); // 2 MB
  float*  Obuf = (float*)ws;                                                 // 4 MB

  const int nblk = (NTOT / 16) / 4;   // 1024 tiles, 4 waves per block
  qkv_project<<<nblk, 128, 0, stream>>>(latent, wq, bq, wk, bk, wv, bv,
                                        Qh, Kh, Vt);
  attn_kernel<<<nblk, 128, 0, stream>>>(Qh, Kh, Vt, Obuf);
  out_project<<<nblk, 128, 0, stream>>>(Obuf, wo, bo, (float*)d_out);
}
